// LAM_Module_19052474925494
// MI455X (gfx1250) — matrix operand order repaired and checked by reading, not compiled, not run on hardware
//
#include <hip/hip_runtime.h>

typedef __attribute__((ext_vector_type(16))) __bf16 v16bf;
typedef __attribute__((ext_vector_type(8)))  float  v8f;

#define D_ELEM   294912     // C*H*W
#define NSEG     10
#define NBATCH   16
#define NC       1280       // N*C  (GEMM M and K)
#define HWL      2304       // H*W  (GEMM N)

static __device__ __forceinline__ unsigned short f2bf(float f) {
    unsigned u = __builtin_bit_cast(unsigned, f);
    u += 0x7FFFu + ((u >> 16) & 1u);           // round-to-nearest-even
    return (unsigned short)(u >> 16);
}
static __device__ __forceinline__ float bf2f(unsigned short h) {
    return __builtin_bit_cast(float, ((unsigned)h) << 16);
}

// ---------------- kernel 0: zero the dot-product accumulators ----------------
__global__ void k_zero(float* p, int n) {
    int t = blockIdx.x * blockDim.x + threadIdx.x;
    if (t < n) p[t] = 0.f;
}

// ------- kernel 1: rolling-window dots: q_n·q_n, q_n·q_{n+1}, q_n·q_{n+2} ----
// Each block owns (b, 4096-elem chunk) and walks all 10 segments keeping a
// 3-row register window, so x is read ~1.2x instead of 3x.
__global__ __launch_bounds__(256) void k_dots(const float* __restrict__ x,
                                              float* __restrict__ dots) {
    const int b = blockIdx.y;
    const int base = blockIdx.x * 4096 + threadIdx.x * 16;
    const int t = threadIdx.x;
    __shared__ float red[768];

    float c0[16], c1[16], c2[16];
    {
        const float* p0 = x + ((size_t)(b * NSEG + 0)) * D_ELEM + base;
        const float* p1 = x + ((size_t)(b * NSEG + 1)) * D_ELEM + base;
        #pragma unroll
        for (int j = 0; j < 4; j++) {
            *(float4*)&c0[4 * j] = *(const float4*)(p0 + 4 * j);
            *(float4*)&c1[4 * j] = *(const float4*)(p1 + 4 * j);
        }
    }
    #pragma unroll
    for (int n = 0; n < NSEG; n++) {
        const float* p2 = x + ((size_t)(b * NSEG + (n + 2) % NSEG)) * D_ELEM + base;
        #pragma unroll
        for (int j = 0; j < 4; j++) *(float4*)&c2[4 * j] = *(const float4*)(p2 + 4 * j);

        float s0 = 0.f, s1 = 0.f, s2 = 0.f;
        #pragma unroll
        for (int j = 0; j < 16; j++) {
            s0 += c0[j] * c0[j];
            s1 += c0[j] * c1[j];
            s2 += c0[j] * c2[j];
        }
        red[t] = s0; red[t + 256] = s1; red[t + 512] = s2;
        __syncthreads();
        for (int s = 128; s > 0; s >>= 1) {
            if (t < s) {
                red[t]       += red[t + s];
                red[t + 256] += red[t + 256 + s];
                red[t + 512] += red[t + 512 + s];
            }
            __syncthreads();
        }
        if (t == 0) {
            atomicAdd(&dots[(b * NSEG + n) * 3 + 0], red[0]);
            atomicAdd(&dots[(b * NSEG + n) * 3 + 1], red[256]);
            atomicAdd(&dots[(b * NSEG + n) * 3 + 2], red[512]);
        }
        __syncthreads();
        #pragma unroll
        for (int j = 0; j < 16; j++) { c0[j] = c1[j]; c1[j] = c2[j]; }
    }
}

// ---------------- kernel 2: softmax -> (alpha,beta,gamma) per (b,n) ----------
__global__ void k_coef(const float* __restrict__ dots,
                       const float* __restrict__ pos_dec,
                       const float* __restrict__ len_dec,
                       float* __restrict__ coef) {
    int t = threadIdx.x;
    if (t >= NBATCH * NSEG) return;
    int n = t % NSEG;
    float pd = pos_dec[n], ld = len_dec[n];
    float d0 = dots[t*3+0], d1 = dots[t*3+1], d2 = dots[t*3+2];
    float l0 = (1.f - pd) * d0 + pd * d1;
    float l1 = ld * ((1.f - pd) * d1 + pd * d2);
    float m  = fmaxf(l0, l1);
    float e0 = __expf(l0 - m), e1 = __expf(l1 - m);
    float inv = 1.f / (e0 + e1);
    float a0 = e0 * inv, a1 = e1 * inv;
    coef[t*3+0] = a0 * (1.f - pd);
    coef[t*3+1] = a0 * pd + a1 * ld * (1.f - pd);
    coef[t*3+2] = a1 * ld * pd;
}

// ---------------- kernel 3: conv_w -> bf16 hi/lo split -----------------------
__global__ void k_wsplit(const float* __restrict__ w,
                         unsigned short* __restrict__ whi,
                         unsigned short* __restrict__ wlo) {
    int t = blockIdx.x * blockDim.x + threadIdx.x;   // covers NC*NC/4
    int i4 = t * 4;
    float4 f = *(const float4*)(w + i4);
    ushort4 hi, lo;
    hi.x = f2bf(f.x); lo.x = f2bf(f.x - bf2f(hi.x));
    hi.y = f2bf(f.y); lo.y = f2bf(f.y - bf2f(hi.y));
    hi.z = f2bf(f.z); lo.z = f2bf(f.z - bf2f(hi.z));
    hi.w = f2bf(f.w); lo.w = f2bf(f.w - bf2f(hi.w));
    *(ushort4*)(whi + i4) = hi;
    *(ushort4*)(wlo + i4) = lo;
}

// -------- kernel 4: feat = a*k + b*k1 + g*k2 -> bf16 hi/lo, TRANSPOSED -------
// Output layout FT[b][hw][nc] (K-contiguous) so GEMM B-staging is plain copies.
// Block: 64c x 64hw tile of one (b, nidx). LDS transpose, coalesced both sides.
__global__ __launch_bounds__(256) void k_feat(
    const float* __restrict__ x, const float* __restrict__ coef,
    unsigned short* __restrict__ fhiT, unsigned short* __restrict__ floT)
{
    constexpr int TP = 72;  // LDS pitch in ushorts for 64-wide hw tile (+8 pad)
    __shared__ __align__(16) unsigned short sHi[64 * TP];
    __shared__ __align__(16) unsigned short sLo[64 * TP];

    const int bx  = blockIdx.x;
    const int hwt = bx % (HWL / 64);          // 36 hw tiles
    const int ct  = (bx / (HWL / 64)) & 1;    // 2 c tiles
    const int bn  = bx / (HWL / 64) / 2;      // b*NSEG+nidx (160)
    const int b   = bn / NSEG, nidx = bn % NSEG;
    const int c0  = ct * 64;
    const int hw0 = hwt * 64;
    const int t   = threadIdx.x;

    const float* c3 = coef + bn * 3;
    const float al = c3[0], be = c3[1], ga = c3[2];

    // phase 1: read x rows (coalesced along hw), combine, split, store LDS [c][hw]
    {
        const int r = t >> 2;                 // 0..63  (c within tile)
        const int q = (t & 3) * 16;           // hw offset within 64
        const size_t roff = (size_t)(c0 + r) * HWL + hw0 + q;
        const float* x0 = x + ((size_t)(b * NSEG + nidx)) * D_ELEM + roff;
        const float* x1 = x + ((size_t)(b * NSEG + (nidx + 1) % NSEG)) * D_ELEM + roff;
        const float* x2 = x + ((size_t)(b * NSEG + (nidx + 2) % NSEG)) * D_ELEM + roff;
        #pragma unroll
        for (int j = 0; j < 4; j++) {
            float4 a = *(const float4*)(x0 + 4 * j);
            float4 u = *(const float4*)(x1 + 4 * j);
            float4 w = *(const float4*)(x2 + 4 * j);
            float f0 = al*a.x + be*u.x + ga*w.x;
            float f1 = al*a.y + be*u.y + ga*w.y;
            float f2 = al*a.z + be*u.z + ga*w.z;
            float f3 = al*a.w + be*u.w + ga*w.w;
            ushort4 hi, lo;
            hi.x = f2bf(f0); lo.x = f2bf(f0 - bf2f(hi.x));
            hi.y = f2bf(f1); lo.y = f2bf(f1 - bf2f(hi.y));
            hi.z = f2bf(f2); lo.z = f2bf(f2 - bf2f(hi.z));
            hi.w = f2bf(f3); lo.w = f2bf(f3 - bf2f(hi.w));
            *(ushort4*)&sHi[r * TP + q + 4 * j] = hi;
            *(ushort4*)&sLo[r * TP + q + 4 * j] = lo;
        }
    }
    __syncthreads();
    // phase 2: gather transposed from LDS, write FT[b][hw][nc] (32B contiguous)
    {
        const int h = t >> 2;                 // 0..63  (hw within tile)
        const int q = (t & 3) * 16;           // c offset within 64
        union { uint4 v[2]; unsigned short s[16]; } oh, ol;
        #pragma unroll
        for (int i = 0; i < 16; i++) {
            oh.s[i] = sHi[(q + i) * TP + h];
            ol.s[i] = sLo[(q + i) * TP + h];
        }
        const size_t o = ((size_t)(b * HWL + hw0 + h)) * NC + nidx * 128 + c0 + q;
        *(uint4*)(fhiT + o)     = oh.v[0];
        *(uint4*)(fhiT + o + 8) = oh.v[1];
        *(uint4*)(floT + o)     = ol.v[0];
        *(uint4*)(floT + o + 8) = ol.v[1];
    }
}

// ---------------- kernel 5: batched GEMM + bias + residual (WMMA bf16x3) -----
__global__ __launch_bounds__(256) void k_gemm(
    const unsigned short* __restrict__ Whi, const unsigned short* __restrict__ Wlo,
    const unsigned short* __restrict__ FhiT, const unsigned short* __restrict__ FloT,
    const float* __restrict__ convb, const float* __restrict__ xres,
    float* __restrict__ out)
{
    constexpr int PITCH = 40;  // 32 bf16 data + 8 pad per LDS row (80B, 16B aligned)
    __shared__ __align__(16) unsigned short sAhi[128 * PITCH];
    __shared__ __align__(16) unsigned short sAlo[128 * PITCH];
    __shared__ __align__(16) unsigned short sBhi[128 * PITCH];  // [n][k]
    __shared__ __align__(16) unsigned short sBlo[128 * PITCH];

    const int n0 = blockIdx.x * 128;
    const int m0 = blockIdx.y * 128;
    const int bb = blockIdx.z;
    const int t    = threadIdx.x;
    const int lane = t & 31;
    const int wv   = t >> 5;
    const int wm   = wv & 1;    // wave M tile (64 rows)
    const int wn   = wv >> 1;   // wave N tile (32 cols)

    v8f acc[4][2];
    #pragma unroll
    for (int i = 0; i < 4; i++)
        #pragma unroll
        for (int j = 0; j < 2; j++)
            #pragma unroll
            for (int e = 0; e < 8; e++) acc[i][j][e] = 0.f;

    const int ar = t >> 1;            // stage row 0..127 (A: m, B: n)
    const int ac = (t & 1) << 4;      // stage k offset 0/16
    const size_t gabase = (size_t)(m0 + ar) * NC + ac;
    const size_t gbbase = ((size_t)(bb * HWL + n0 + ar)) * NC + ac;

    // register-staged prefetch (tile k+1 loads overlap tile-k WMMAs)
    uint4 rah0, rah1, ral0, ral1, rbh0, rbh1, rbl0, rbl1;
    {
        rah0 = *(const uint4*)(Whi  + gabase);  rah1 = *(const uint4*)(Whi  + gabase + 8);
        ral0 = *(const uint4*)(Wlo  + gabase);  ral1 = *(const uint4*)(Wlo  + gabase + 8);
        rbh0 = *(const uint4*)(FhiT + gbbase);  rbh1 = *(const uint4*)(FhiT + gbbase + 8);
        rbl0 = *(const uint4*)(FloT + gbbase);  rbl1 = *(const uint4*)(FloT + gbbase + 8);
    }

    for (int kt = 0; kt < NC / 32; kt++) {
        // ---- store staged registers to LDS ----
        *(uint4*)&sAhi[ar * PITCH + ac]     = rah0;
        *(uint4*)&sAhi[ar * PITCH + ac + 8] = rah1;
        *(uint4*)&sAlo[ar * PITCH + ac]     = ral0;
        *(uint4*)&sAlo[ar * PITCH + ac + 8] = ral1;
        *(uint4*)&sBhi[ar * PITCH + ac]     = rbh0;
        *(uint4*)&sBhi[ar * PITCH + ac + 8] = rbh1;
        *(uint4*)&sBlo[ar * PITCH + ac]     = rbl0;
        *(uint4*)&sBlo[ar * PITCH + ac + 8] = rbl1;
        __syncthreads();

        // ---- prefetch next tile from global (latency hidden under WMMAs) ----
        if (kt + 1 < NC / 32) {
            const size_t ga = gabase + (kt + 1) * 32;
            const size_t gb = gbbase + (kt + 1) * 32;
            rah0 = *(const uint4*)(Whi  + ga);  rah1 = *(const uint4*)(Whi  + ga + 8);
            ral0 = *(const uint4*)(Wlo  + ga);  ral1 = *(const uint4*)(Wlo  + ga + 8);
            rbh0 = *(const uint4*)(FhiT + gb);  rbh1 = *(const uint4*)(FhiT + gb + 8);
            rbl0 = *(const uint4*)(FloT + gb);  rbl1 = *(const uint4*)(FloT + gb + 8);
        }

        // ---- load fragments per ISA layouts ----
        v16bf ah[4], alo[4], bh[2], bl[2];
        const int lr = lane & 15;
        const int lk = (lane >> 4) << 3;   // A: K half-select (0/8)
        #pragma unroll
        for (int mi = 0; mi < 4; mi++) {
            const unsigned short* p = &sAhi[(wm * 64 + mi * 16 + lr) * PITCH + lk];
            union { uint4 q[2]; v16bf v; } u;
            u.q[0] = *(const uint4*)p; u.q[1] = *(const uint4*)(p + 16);
            ah[mi] = u.v;
            const unsigned short* pl = &sAlo[(wm * 64 + mi * 16 + lr) * PITCH + lk];
            union { uint4 q[2]; v16bf v; } u2;
            u2.q[0] = *(const uint4*)pl; u2.q[1] = *(const uint4*)(pl + 16);
            alo[mi] = u2.v;
        }
        const int kb = (lane >> 4) << 3;   // B: K half-select (0/8)
        #pragma unroll
        for (int ni = 0; ni < 2; ni++) {
            const unsigned short* p = &sBhi[(wn * 32 + ni * 16 + lr) * PITCH + kb];
            union { uint4 q[2]; v16bf v; } u;
            u.q[0] = *(const uint4*)p; u.q[1] = *(const uint4*)(p + 16);
            bh[ni] = u.v;
            const unsigned short* pl = &sBlo[(wn * 32 + ni * 16 + lr) * PITCH + kb];
            union { uint4 q[2]; v16bf v; } u2;
            u2.q[0] = *(const uint4*)pl; u2.q[1] = *(const uint4*)(pl + 16);
            bl[ni] = u2.v;
        }

        // ---- bf16x3: hi*hi + hi*lo + lo*hi, f32 accumulate ----
        #pragma unroll
        for (int mi = 0; mi < 4; mi++)
            #pragma unroll
            for (int ni = 0; ni < 2; ni++) {
                acc[mi][ni] = __builtin_amdgcn_wmma_f32_16x16x32_bf16(
                    false, ah[mi],  false, bh[ni], (short)0, acc[mi][ni], false, false);
                acc[mi][ni] = __builtin_amdgcn_wmma_f32_16x16x32_bf16(
                    false, ah[mi],  false, bl[ni], (short)0, acc[mi][ni], false, false);
                acc[mi][ni] = __builtin_amdgcn_wmma_f32_16x16x32_bf16(
                    false, alo[mi], false, bh[ni], (short)0, acc[mi][ni], false, false);
            }
        __syncthreads();
    }

    // ---- epilogue: bias + residual, C layout (VGPR v -> M = v + 8*(lane/16)) ----
    const int lm = (lane >> 4) * 8;
    const int ln = lane & 15;
    #pragma unroll
    for (int mi = 0; mi < 4; mi++) {
        #pragma unroll
        for (int v = 0; v < 8; v++) {
            const int m = m0 + wm * 64 + mi * 16 + lm + v;
            const float bias = convb[m];
            #pragma unroll
            for (int ni = 0; ni < 2; ni++) {
                const int n = n0 + wn * 32 + ni * 16 + ln;
                const size_t o = ((size_t)(bb * NC + m)) * HWL + n;
                out[o] = acc[mi][ni][v] + bias + xres[o];
            }
        }
    }
}

extern "C" void kernel_launch(void* const* d_in, const int* in_sizes, int n_in,
                              void* d_out, int out_size, void* d_ws, size_t ws_size,
                              hipStream_t stream) {
    (void)in_sizes; (void)n_in; (void)out_size; (void)ws_size;
    const float* x       = (const float*)d_in[0];
    const float* pos_dec = (const float*)d_in[1];
    const float* len_dec = (const float*)d_in[2];
    const float* conv_w  = (const float*)d_in[3];
    const float* conv_b  = (const float*)d_in[4];
    float* out = (float*)d_out;

    char* ws = (char*)d_ws;
    float* dots = (float*)(ws);                    // 480 f32
    float* coef = (float*)(ws + 2048);             // 480 f32
    unsigned short* Whi  = (unsigned short*)(ws + 4096);
    unsigned short* Wlo  = Whi  + (size_t)NC * NC;
    unsigned short* FhiT = Wlo  + (size_t)NC * NC;
    unsigned short* FloT = FhiT + (size_t)NBATCH * HWL * NC;

    k_zero<<<2, 256, 0, stream>>>(dots, NBATCH * NSEG * 3);
    k_dots<<<dim3(D_ELEM / 4096, NBATCH), 256, 0, stream>>>(x, dots);
    k_coef<<<1, 256, 0, stream>>>(dots, pos_dec, len_dec, coef);
    k_wsplit<<<(NC * NC / 4) / 256, 256, 0, stream>>>(conv_w, Whi, Wlo);
    k_feat<<<NBATCH * NSEG * 2 * (HWL / 64), 256, 0, stream>>>(x, coef, FhiT, FloT);
    k_gemm<<<dim3(HWL / 128, NC / 128, NBATCH), 256, 0, stream>>>(Whi, Wlo, FhiT, FloT,
                                                                  conv_b, x, out);
}
